// AlignmentContrastiveLoss_82626580841015
// MI455X (gfx1250) — compile-verified, matrix operand order checked by reading, not run on hardware
//
#include <hip/hip_runtime.h>

// ---------------------------------------------------------------------------
// AlignmentContrastiveLoss for MI455X (gfx1250, wave32, WMMA bf16 16x16x32)
//
// Stage 1: each workgroup owns a 2x2 block of (b,t) pairs -> a 128x128 output
//   macro-tile over K=1024, pipelined (ping-pong LDS), fused mask+colmax+sum.
//   8 waves, each computing a 32x64 block = 8 v_wmma_f32_16x16x32_bf16/chunk.
// Stage 2: 128x128 hinge-loss reduction to a single scalar.
// ---------------------------------------------------------------------------

typedef __attribute__((ext_vector_type(16))) __bf16 v16bf;
typedef __attribute__((ext_vector_type(8)))  __bf16 v8bf;
typedef __attribute__((ext_vector_type(8)))  float  v8f;

#define MARGIN 0.2f

constexpr int B      = 128;   // batch
constexpr int LI_RAW = 65;    // raw image-set length
constexpr int LS_RAW = 67;    // raw sentence length
constexpr int D      = 1024;  // feature dim
constexpr int LI     = 64;    // after [:,1:,:]
constexpr int LS     = 64;    // after [:,1:-2,:]
constexpr int KC     = 32;    // bf16 WMMA K per step

__global__ __launch_bounds__(256)
void aggr_kernel(const float* __restrict__ im_set,
                 const float* __restrict__ s_seq,
                 const int*   __restrict__ im_len,
                 const int*   __restrict__ s_len,
                 float*       __restrict__ aggr)
{
    // ping-pong bf16 slabs: 128 rows (b0|b1 resp. t0|t1) x 32 K
    __shared__ __bf16 sA[2][128 * KC];
    __shared__ __bf16 sB[2][128 * KC];
    __shared__ float  lC[LI * LS];     // one pair's masked 64x64 tile
    __shared__ float  colmax[LS];

    const int t0   = blockIdx.x * 2;   // sentence pair (t0, t0+1)
    const int b0   = blockIdx.y * 2;   // image pair    (b0, b0+1)
    const int tid  = threadIdx.x;
    const int wave = tid >> 5;         // 0..7
    const int lane = tid & 31;
    const int half = lane >> 4;
    const int mn   = lane & 15;

    // wave tiling of the 128x128 macro-tile: 32 rows x 64 cols per wave
    const int row_grp = wave >> 1;     // 0..3  -> rows [32*row_grp, +32)
    const int col_grp = wave & 1;      // 0..1  -> cols [64*col_grp, +64)

    // staging coords: each thread stages 16 consecutive K of one slab row
    const int srow = tid >> 1;         // 0..127
    const int scol = (tid & 1) * 16;   // 0 or 16
    const int lrow = srow & 63;
    const float* gA = im_set +
        ((size_t)((srow < 64) ? b0 : b0 + 1) * LI_RAW + 1 + lrow) * D + scol;
    const float* gB = s_seq +
        ((size_t)((srow < 64) ? t0 : t0 + 1) * LS_RAW + 1 + lrow) * D + scol;

    float4 rA[4], rB[4];

    auto ldg = [&](int kk) {
        const float4* pa = (const float4*)(gA + kk);
        const float4* pb = (const float4*)(gB + kk);
        rA[0] = pa[0]; rA[1] = pa[1]; rA[2] = pa[2]; rA[3] = pa[3];
        rB[0] = pb[0]; rB[1] = pb[1]; rB[2] = pb[2]; rB[3] = pb[3];
    };
    auto stg = [&](int buf) {
        v8bf ha0 = { (__bf16)rA[0].x, (__bf16)rA[0].y, (__bf16)rA[0].z, (__bf16)rA[0].w,
                     (__bf16)rA[1].x, (__bf16)rA[1].y, (__bf16)rA[1].z, (__bf16)rA[1].w };
        v8bf ha1 = { (__bf16)rA[2].x, (__bf16)rA[2].y, (__bf16)rA[2].z, (__bf16)rA[2].w,
                     (__bf16)rA[3].x, (__bf16)rA[3].y, (__bf16)rA[3].z, (__bf16)rA[3].w };
        v8bf hb0 = { (__bf16)rB[0].x, (__bf16)rB[0].y, (__bf16)rB[0].z, (__bf16)rB[0].w,
                     (__bf16)rB[1].x, (__bf16)rB[1].y, (__bf16)rB[1].z, (__bf16)rB[1].w };
        v8bf hb1 = { (__bf16)rB[2].x, (__bf16)rB[2].y, (__bf16)rB[2].z, (__bf16)rB[2].w,
                     (__bf16)rB[3].x, (__bf16)rB[3].y, (__bf16)rB[3].z, (__bf16)rB[3].w };
        *(v8bf*)&sA[buf][srow * KC + scol]     = ha0;
        *(v8bf*)&sA[buf][srow * KC + scol + 8] = ha1;
        *(v8bf*)&sB[buf][srow * KC + scol]     = hb0;
        *(v8bf*)&sB[buf][srow * KC + scol + 8] = hb1;
    };

    union Frag { v16bf v; v8bf h[2]; };
    v8f acc[2][4] = {};

    // pipeline prologue: stage chunk 0
    ldg(0);
    stg(0);
    __syncthreads();

    for (int kk = 0; kk < D; kk += KC) {
        const int  cur  = (kk >> 5) & 1;
        const bool more = (kk + KC) < D;
        if (more) ldg(kk + KC);        // overlap next-chunk global loads

        // fragments per the 16-bit A/B VGPR layout: two contiguous 16B runs
        Frag fa[2], fb[4];
        #pragma unroll
        for (int ia = 0; ia < 2; ++ia) {
            const __bf16* p = &sA[cur][((row_grp * 2 + ia) * 16 + mn) * KC + half * 8];
            fa[ia].h[0] = *(const v8bf*)p;
            fa[ia].h[1] = *(const v8bf*)(p + 16);
        }
        #pragma unroll
        for (int jb = 0; jb < 4; ++jb) {
            const __bf16* p = &sB[cur][((col_grp * 4 + jb) * 16 + mn) * KC + half * 8];
            fb[jb].h[0] = *(const v8bf*)p;
            fb[jb].h[1] = *(const v8bf*)(p + 16);
        }

        #pragma unroll
        for (int ia = 0; ia < 2; ++ia)
            #pragma unroll
            for (int jb = 0; jb < 4; ++jb)
                acc[ia][jb] = __builtin_amdgcn_wmma_f32_16x16x32_bf16(
                    false, fa[ia].v, false, fb[jb].v,
                    (short)0, acc[ia][jb], false, false);

        if (more) stg(cur ^ 1);        // fill the other buffer
        __syncthreads();
    }

    // ---- epilogue: 4 pairs sequentially through the shared 64x64 buffer
    for (int p = 0; p < 4; ++p) {
        const int pb = p >> 1, pt = p & 1;
        if ((row_grp >> 1) == pb && col_grp == pt) {
            const int im_l = im_len[b0 + pb] - 1;
            const int s_l  = s_len[t0 + pt] - 3;
            #pragma unroll
            for (int ia = 0; ia < 2; ++ia)
                #pragma unroll
                for (int jb = 0; jb < 4; ++jb)
                    #pragma unroll
                    for (int v = 0; v < 8; ++v) {
                        const int row = (row_grp & 1) * 32 + ia * 16 + v + 8 * half;
                        const int col = jb * 16 + mn;
                        lC[row * LS + col] =
                            (row < im_l && col < s_l) ? acc[ia][jb][v] : 0.0f;
                    }
        }
        __syncthreads();
        // aggr[b,t] = sum_j max_i lC[i][j] (zeros participate, as in ref)
        if (tid < LS) {
            float m = lC[tid];
            #pragma unroll 8
            for (int r = 1; r < LI; ++r) m = fmaxf(m, lC[r * LS + tid]);
            colmax[tid] = m;
        }
        __syncthreads();
        if (tid == 0) {
            float s = 0.0f;
            #pragma unroll 8
            for (int j = 0; j < LS; ++j) s += colmax[j];
            aggr[(size_t)(b0 + pb) * B + (t0 + pt)] = s;
        }
        // next pair's colmax writes are gated by the wait at its first barrier
    }
}

__global__ __launch_bounds__(128)
void loss_kernel(const float* __restrict__ aggr, float* __restrict__ out)
{
    __shared__ float red[B];
    const int tid = threadIdx.x;

    const float db = aggr[(size_t)tid * B + tid];   // diag[tid]

    // row-max of cost_s (b=tid) and col-max of cost_im (t=tid); both use
    // diag[tid]; eye entries excluded (max starts at 0, matching the
    // zeroed diagonal in the reference).
    float mrow = 0.0f, mcol = 0.0f;
    for (int k = 0; k < B; ++k) {
        if (k == tid) continue;
        mrow = fmaxf(mrow, fmaxf(MARGIN + aggr[(size_t)tid * B + k] - db, 0.0f));
        mcol = fmaxf(mcol, fmaxf(MARGIN + aggr[(size_t)k * B + tid] - db, 0.0f));
    }
    red[tid] = mrow + mcol;
    __syncthreads();

    for (int s = 64; s > 0; s >>= 1) {
        if (tid < s) red[tid] += red[tid + s];
        __syncthreads();
    }
    if (tid == 0) out[0] = red[0];
}

extern "C" void kernel_launch(void* const* d_in, const int* in_sizes, int n_in,
                              void* d_out, int out_size, void* d_ws, size_t ws_size,
                              hipStream_t stream)
{
    const float* im_set = (const float*)d_in[0];
    const float* s_seq  = (const float*)d_in[1];
    const int*   im_len = (const int*)d_in[2];
    const int*   s_len  = (const int*)d_in[3];

    float* aggr = (float*)d_ws;            // 128*128*4 = 64 KB scratch

    dim3 grid(B / 2, B / 2);               // 2x2 (b,t) pairs per workgroup
    aggr_kernel<<<grid, 256, 0, stream>>>(im_set, s_seq, im_len, s_len, aggr);
    loss_kernel<<<1, B, 0, stream>>>(aggr, (float*)d_out);
}
